// FourierLayer_27075473834703
// MI455X (gfx1250) — matrix operand order repaired and checked by reading, not compiled, not run on hardware
//
#include <hip/hip_runtime.h>
#include <hip/hip_bf16.h>

// ---------------- types ----------------
typedef _Float16      v16h  __attribute__((ext_vector_type(16)));
typedef float         v8f   __attribute__((ext_vector_type(8)));
typedef unsigned int  uv4   __attribute__((ext_vector_type(4)));
typedef unsigned int  u32x4 __attribute__((ext_vector_type(4)));
typedef int           i32x4 __attribute__((ext_vector_type(4)));
typedef int           i32x8 __attribute__((ext_vector_type(8)));

union Frag { v16h h; uv4 q[2]; };

#define NPERSEG 256
#define NFREQS  129
#define NTILES_F 9              // 9 cos tiles (144 cols) + 9 sin tiles
#define ROWLEN  8192
#define TWO_PI_OVER_N (6.283185307179586f / 256.0f)

// ---------------- kernel 0: zero output ----------------
__global__ void welch_zero(float* __restrict__ out, int n) {
    int i = blockIdx.x * blockDim.x + threadIdx.x;
    if (i < n) out[i] = 0.0f;
}

// ---------------- kernel 1: build f16 DFT basis in WMMA-B fragment layout ----
// Btab[((nt*8+kt)*32 + lane)*16 + j] : lane<16 -> N=lane, K=kt*32 + j
//                                      lane>=16 -> N=lane-16, K=kt*32+16+j
// nt in 0..8 : cos(2*pi*kf*K/256), kf = nt*16 + N  (zero for kf>128)
// nt in 9..17: -sin(...)          , kf = (nt-9)*16 + N
__global__ void welch_build_btab(unsigned short* __restrict__ Btab) {
    int gid = blockIdx.x * blockDim.x + threadIdx.x;
    if (gid >= 18 * 8 * 32 * 16) return;
    int j  = gid & 15;
    int ln = (gid >> 4) & 31;
    int kt = (gid >> 9) & 7;
    int nt = gid >> 12;
    int K  = kt * 32 + ((ln >> 4) * 16) + j;
    int N  = ln & 15;
    float val = 0.0f;
    if (nt < NTILES_F) {
        int kf = nt * 16 + N;
        if (kf <= 128) {
            int ph = (kf * K) & 255;
            val = __cosf((float)ph * TWO_PI_OVER_N);
        }
    } else {
        int kf = (nt - NTILES_F) * 16 + N;
        if (kf <= 128) {
            int ph = (kf * K) & 255;
            val = -__sinf((float)ph * TWO_PI_OVER_N);
        }
    }
    _Float16 hv = (_Float16)val;
    Btab[gid] = __builtin_bit_cast(unsigned short, hv);
}

// ---------------- kernel 2: main Welch kernel -----------------------
// grid = 2 * nrows, block = 256 (8 waves). Block (r, hb) handles segments
// hb*32 .. hb*32+31 of row r (hb==1: 31 real + 1 zero-pad segment).
__global__ __launch_bounds__(256) void welch_main(
        const float* __restrict__ X,
        const unsigned short* __restrict__ Btab,
        float* __restrict__ out) {

    __shared__ float        rowBuf[4352];       // up to 4224 f32 samples
    __shared__ unsigned int Abuf[32 * 132];     // 32 segs x 256 f16 (+16B pad/row)
    __shared__ float        wWin[256];
    __shared__ float        meansS[32];

    const int t    = threadIdx.x;
    const int ln   = t & 31;
    const int hlf  = ln >> 4;
    const int wid  = __builtin_amdgcn_readfirstlane(t >> 5);   // wave id, scalar
    const int blk  = blockIdx.x;
    const int r    = blk >> 1;
    const int hb   = blk & 1;
    const int validSegs = hb ? 31 : 32;
    const int L         = hb ? 4096 : 4224;     // floats staged in LDS
    const float* rowPtr = X + (size_t)r * ROWLEN + (size_t)hb * 4096;

    // Hann window table
    wWin[t] = 0.5f - 0.5f * __cosf((float)t * TWO_PI_OVER_N);

    // --- TDM: async 1-D tensor load of the half-row into LDS (wave 0 only) ---
    if (wid == 0) {
        unsigned lds_off = (unsigned)(size_t)(&rowBuf[0]);   // low 32 bits = LDS offset
        unsigned long long ga = (unsigned long long)rowPtr;
        u32x4 g0 = (u32x4)0u;
        g0[0] = 1u;                                  // count=1, no gather
        g0[1] = lds_off;                             // lds_addr
        g0[2] = (unsigned)(ga & 0xffffffffull);      // global_addr[31:0]
        g0[3] = (unsigned)(ga >> 32) | (2u << 30);   // global_addr[56:32] | type=2
        i32x8 g1 = (i32x8)0;
        g1[0] = 0x00020000;                          // data_size=2 (4B), mask=0
        g1[1] = (int)((unsigned)(L & 0xffff) << 16); // tensor_dim0[15:0]
        g1[2] = (int)((unsigned)(L >> 16) | (1u << 16)); // dim0 hi | tensor_dim1=1
        g1[3] = (int)((unsigned)L << 16);            // tile_dim0 = L
        g1[4] = 1;                                   // tile_dim1 = 1
        g1[5] = L;                                   // tensor_dim0_stride lo
        i32x4 g2 = (i32x4)0;
        i32x4 g3 = (i32x4)0;
        i32x8 g4 = (i32x8)0;                         // extra group (6-arg toolchain)
        __builtin_amdgcn_tensor_load_to_lds(g0, g1, g2, g3, g4, 0);
        __builtin_amdgcn_s_wait_tensorcnt(0);
    }
    __syncthreads();   // rowBuf + wWin visible to all waves

    // --- per-segment means (for k=0/1 Hann fix-up): 8 threads per segment ---
    {
        int s = t >> 3, p = t & 7;
        float acc = 0.0f;
        if (s < validSegs) {
            int base = s * 128 + p * 32;
            #pragma unroll
            for (int i = 0; i < 32; ++i) acc += rowBuf[base + i];
        }
        acc += __shfl_down(acc, 4, 8);
        acc += __shfl_down(acc, 2, 8);
        acc += __shfl_down(acc, 1, 8);
        if (p == 0) meansS[s] = acc * (1.0f / 256.0f);
    }

    // --- build windowed f16 A panel: Abuf[s][n] = f16(row[s*128+n] * w[n]) ---
    #pragma unroll
    for (int i = 0; i < 16; ++i) {
        int e2 = t + 256 * i;            // dword element 0..4095
        int s  = e2 >> 7;
        int n2 = e2 & 127;
        int n  = n2 * 2;
        float x0 = 0.0f, x1 = 0.0f;
        if (s < validSegs) {
            x0 = rowBuf[s * 128 + n];
            x1 = rowBuf[s * 128 + n + 1];
        }
        _Float16 a0 = (_Float16)(x0 * wWin[n]);
        _Float16 a1 = (_Float16)(x1 * wWin[n + 1]);
        unsigned u = (unsigned)__builtin_bit_cast(unsigned short, a0)
                   | ((unsigned)__builtin_bit_cast(unsigned short, a1) << 16);
        Abuf[s * 132 + n2] = u;
    }
    __syncthreads();

    // --- WMMA GEMM: 18 (M-tile, freq-tile) pairs over 8 waves ----------------
    const float invScale = (1.0f / 96.0f) / 63.0f;   // 1/sum(w^2) / nseg
    for (int p = wid; p < 2 * NTILES_F; p += 8) {
        int mt  = p / NTILES_F;
        int ntc = p % NTILES_F;
        v8f cr = {0.f,0.f,0.f,0.f,0.f,0.f,0.f,0.f};
        v8f ci = {0.f,0.f,0.f,0.f,0.f,0.f,0.f,0.f};
        const unsigned int* abase = &Abuf[(mt * 16 + (ln & 15)) * 132];
        #pragma unroll
        for (int kt = 0; kt < 8; ++kt) {
            Frag fa, fbc, fbs;
            const uv4* pa = (const uv4*)(abase + kt * 16 + hlf * 8);
            fa.q[0] = pa[0];
            fa.q[1] = pa[1];                          // +16B: K 8..15 of this lane half
            const uv4* pbc = (const uv4*)(Btab + (size_t)(((ntc * 8 + kt) * 32 + ln) * 16));
            fbc.q[0] = pbc[0]; fbc.q[1] = pbc[1];
            const uv4* pbs = (const uv4*)(Btab + (size_t)((((ntc + NTILES_F) * 8 + kt) * 32 + ln) * 16));
            fbs.q[0] = pbs[0]; fbs.q[1] = pbs[1];
            cr = __builtin_amdgcn_wmma_f32_16x16x32_f16(false, fa.h, false, fbc.h,
                                                        (short)0, cr, false, false);
            ci = __builtin_amdgcn_wmma_f32_16x16x32_f16(false, fa.h, false, fbs.h,
                                                        (short)0, ci, false, false);
        }
        // epilogue: mean fix-up (Hann: W[0]=128, W[1]=-64), power, seg-sum
        int kf = ntc * 16 + (ln & 15);
        float psum = 0.0f;
        #pragma unroll
        for (int v = 0; v < 8; ++v) {
            float m  = meansS[mt * 16 + v + hlf * 8];
            float Fr = cr[v];
            float Fi = ci[v];
            if (kf == 0)      Fr -= 128.0f * m;
            else if (kf == 1) Fr += 64.0f * m;
            psum += Fr * Fr + Fi * Fi;
        }
        psum += __shfl_xor(psum, 16, 32);             // combine M halves (same kf)
        if (hlf == 0 && kf <= 128) {
            float coef = ((kf == 0) || (kf == 128)) ? invScale : 2.0f * invScale;
            atomicAdd(&out[(size_t)r * NFREQS + kf], psum * coef);
        }
    }
}

// ---------------- host entry ----------------
extern "C" void kernel_launch(void* const* d_in, const int* in_sizes, int n_in,
                              void* d_out, int out_size, void* d_ws, size_t ws_size,
                              hipStream_t stream) {
    const float* X = (const float*)d_in[0];
    float* out = (float*)d_out;
    unsigned short* Btab = (unsigned short*)d_ws;     // 18*8*32*16 f16 = 147456 B

    welch_zero<<<(out_size + 255) / 256, 256, 0, stream>>>(out, out_size);

    const int btab_elems = 18 * 8 * 32 * 16;
    welch_build_btab<<<(btab_elems + 255) / 256, 256, 0, stream>>>(Btab);

    int rows = in_sizes[0] / ROWLEN;                  // 4096
    welch_main<<<rows * 2, 256, 0, stream>>>(X, Btab, out);
}
